// BigramEncoder_65343632441716
// MI455X (gfx1250) — matrix operand order repaired and checked by reading, not compiled, not run on hardware
//
#include <hip/hip_runtime.h>
#include <cstdint>

// ---------------------------------------------------------------------------
// Types for CDNA5 WMMA (gfx1250, wave32)
// ---------------------------------------------------------------------------
typedef __attribute__((ext_vector_type(16))) __bf16 v16bf;
typedef __attribute__((ext_vector_type(8)))  float  v8f;

typedef unsigned short u16;
typedef unsigned int   u32;
typedef long long      ll;

__device__ __forceinline__ u16 f32_to_bf16(float f) {
    u32 u = __builtin_bit_cast(u32, f);
    u32 r = (u + 0x7FFFu + ((u >> 16) & 1u)) >> 16;   // round-to-nearest-even
    return (u16)r;
}

// CDNA5 async memory->LDS copy (ASYNCcnt-tracked). INST_OFFSET applies to both
// the LDS and the global address (ISA 08_async_tensor §4.4), so one address
// pair covers two 16B chunks via offset:0 / offset:16.
__device__ __forceinline__ void async_copy_b128(u32 lds_off, const u16* gptr) {
    asm volatile("global_load_async_to_lds_b128 %0, %1, off"
                 :: "v"(lds_off), "v"(gptr) : "memory");
}
__device__ __forceinline__ void async_copy_b128_o16(u32 lds_off, const u16* gptr) {
    asm volatile("global_load_async_to_lds_b128 %0, %1, off offset:16"
                 :: "v"(lds_off), "v"(gptr) : "memory");
}
__device__ __forceinline__ void wait_async0() {
    asm volatile("s_wait_asynccnt 0x0" ::: "memory");
}
__device__ __forceinline__ u32 lds_off32(const void* p) {
    // flat LDS addresses keep the wave-local LDS offset in addr[31:0]
    return (u32)(uintptr_t)p;
}

// ---------------------------------------------------------------------------
// f32 -> bf16 conversion
// ---------------------------------------------------------------------------
__global__ void cvt_f32_bf16_kernel(const float* __restrict__ src,
                                    u16* __restrict__ dst, ll n) {
    ll i = (ll)blockIdx.x * blockDim.x + threadIdx.x;
    if (i < n) dst[i] = f32_to_bf16(src[i]);
}

// ---------------------------------------------------------------------------
// Generic batched bf16 GEMM, f32 accumulate, WMMA 16x16x32, double-buffered
// LDS with async global->LDS staging.
//   D[b, m, n] = scale * sum_k A[b, m, k] * B[b, k, n]  (+ bias)
// A row-major (lda). B element (k,n) at Bg + b*sBb + k*ldbK + n*ldbN.
//   BCOL=true : ldbK == 1 (k contiguous)  -> async b128 straight into Bs[n][k]
//   BCOL=false: ldbN == 1 (n contiguous)  -> register-staged transpose scatter
// Block tile 128x128, K-step 32, 8 waves of 32 lanes, wave tile 32x64.
// ---------------------------------------------------------------------------
#define BM 128
#define BN 128
#define BK 32
#define LDSS 40   // padded row stride in u16 (80B: 16B-aligned, conflict-skewed)

template<bool BCOL>
__global__ __launch_bounds__(256)
void gemm_bf16_wmma_kernel(const u16* __restrict__ Ag, ll sAb, int lda,
                           const u16* __restrict__ Bg, ll sBb, int ldbK, int ldbN,
                           float* __restrict__ Df, u16* __restrict__ Dh,
                           ll sDb, int ldd,
                           int K, float scale,
                           const float* __restrict__ bias, int biasMode)
{
    __shared__ u16 As[2][BM][LDSS];   // [buf][m][k]
    __shared__ u16 Bs[2][BN][LDSS];   // [buf][n][k]  (transposed)

    const int tid   = threadIdx.x;
    const int lane  = tid & 31;
    const int wave  = tid >> 5;
    const int waveM = wave >> 1;        // 0..3  -> 32-row slab
    const int waveN = wave & 1;         // 0..1  -> 64-col slab
    const int lhi   = lane >> 4;        // lane half (WMMA layout)
    const int llo   = lane & 15;

    const int bM = blockIdx.y * BM;
    const int bN = blockIdx.x * BN;

    Ag += (ll)blockIdx.z * sAb;
    Bg += (ll)blockIdx.z * sBb;

    // per-thread staging coordinates (32 bytes = 2 x b128 per thread per tile)
    const int rowA = tid >> 1;                 // 0..127
    const int colA = (tid & 1) * 16;           // 0 or 16
    const int nB   = tid >> 1;                 // BCOL: B column (n)
    const int kB   = (tid & 1) * 16;           // BCOL: k base
    const int kS   = tid >> 3;                 // !BCOL: k row
    const int nS   = (tid & 7) * 16;           // !BCOL: n base (16 wide)

    const v8f vzero = {0.f,0.f,0.f,0.f,0.f,0.f,0.f,0.f};
    v8f acc[2][4];
#pragma unroll
    for (int i = 0; i < 2; ++i)
#pragma unroll
        for (int j = 0; j < 4; ++j) acc[i][j] = vzero;

    union AFrag { v16bf v; u32 u[8]; };
    union BFrag { v16bf v; u32 u[8]; };

    // stage tile starting at k0 into buffer nb
    auto stage = [&](int k0, int nb) {
        // A tile (128 x 32): async, 32B per thread
        {
            const u16* src = Ag + (ll)(bM + rowA) * lda + (k0 + colA);
            u32 dst = lds_off32(&As[nb][rowA][colA]);
            async_copy_b128(dst, src);
            async_copy_b128_o16(dst, src);
        }
        if (BCOL) {
            // k contiguous: async straight into transposed LDS tile
            const u16* src = Bg + (ll)(bN + nB) * ldbN + (k0 + kB);
            u32 dst = lds_off32(&Bs[nb][nB][kB]);
            async_copy_b128(dst, src);
            async_copy_b128_o16(dst, src);
        } else {
            // n contiguous: register-stage 16 elems, scatter-transpose to LDS
            const u16* src = Bg + (ll)(k0 + kS) * ldbK + (bN + nS);
            uint4 d0 = *reinterpret_cast<const uint4*>(src);
            uint4 d1 = *reinterpret_cast<const uint4*>(src + 8);
            u16 tmp[16];
            *reinterpret_cast<uint4*>(tmp)     = d0;
            *reinterpret_cast<uint4*>(tmp + 8) = d1;
#pragma unroll
            for (int j = 0; j < 16; ++j) Bs[nb][nS + j][kS] = tmp[j];
        }
    };

    // prologue: fill buffer 0
    stage(0, 0);
    wait_async0();
    __syncthreads();

    int buf = 0;
    for (int k0 = 0; k0 < K; k0 += BK) {
        const int nb = buf ^ 1;
        if (k0 + BK < K) stage(k0 + BK, nb);   // overlap next-tile staging

        // ---- build fragments (ISA 7.12.2 16-bit layouts) -------------------
        AFrag af[2];
#pragma unroll
        for (int tm = 0; tm < 2; ++tm) {
            int r = waveM * 32 + tm * 16 + llo;      // A row (M), lane = M
#pragma unroll
            for (int j = 0; j < 4; ++j) {
                int kA = lhi * 8 + 2 * j;            // K pair base per lane-half
                af[tm].u[j]     = *reinterpret_cast<const u32*>(&As[buf][r][kA]);
                af[tm].u[4 + j] = *reinterpret_cast<const u32*>(&As[buf][r][16 + kA]);
            }
        }
        BFrag bfr[4];
#pragma unroll
        for (int tn = 0; tn < 4; ++tn) {
            int ncol = waveN * 64 + tn * 16 + llo;   // B col (N), lane = N
            int kb   = lhi * 8;                      // K 8-group base per lane-half (table order)
#pragma unroll
            for (int j = 0; j < 8; ++j)
                bfr[tn].u[j] =
                    *reinterpret_cast<const u32*>(&Bs[buf][ncol][kb + (j >> 2) * 16 + 2 * (j & 3)]);
        }

        // ---- 8 WMMAs per K-step per wave -----------------------------------
#pragma unroll
        for (int tm = 0; tm < 2; ++tm)
#pragma unroll
            for (int tn = 0; tn < 4; ++tn)
                acc[tm][tn] = __builtin_amdgcn_wmma_f32_16x16x32_bf16(
                    false, af[tm].v, false, bfr[tn].v,
                    (short)0, acc[tm][tn], false, false);

        wait_async0();       // our async writes into nb are done
        __syncthreads();     // everyone finished reading buf / writing nb
        buf = nb;
    }

    // ---- epilogue: D layout VGPR r -> M = r + 8*lanehalf, N = lane&15 ------
    union DFrag { v8f v; float f[8]; };
#pragma unroll
    for (int tm = 0; tm < 2; ++tm) {
#pragma unroll
        for (int tn = 0; tn < 4; ++tn) {
            DFrag d; d.v = acc[tm][tn];
            int col     = bN + waveN * 64 + tn * 16 + llo;
            int rowBase = bM + waveM * 32 + tm * 16 + lhi * 8;
            float bn_ = (biasMode == 2) ? bias[col] : 0.f;
#pragma unroll
            for (int r = 0; r < 8; ++r) {
                int row = rowBase + r;
                float v = d.f[r] * scale + bn_;
                if (biasMode == 1) v += bias[row];
                ll off = (ll)blockIdx.z * sDb + (ll)row * ldd + col;
                if (Df) Df[off] = v;
                if (Dh) Dh[off] = f32_to_bf16(v);
            }
        }
    }
}

// ---------------------------------------------------------------------------
// Per-token max over node axis: amax[b,t] = max_n sim[b,n,t]
// ---------------------------------------------------------------------------
__global__ void colmax_kernel(const float* __restrict__ sim,
                              float* __restrict__ amax, int Nn, int T) {
    int idx = blockIdx.x * blockDim.x + threadIdx.x;   // b*T + t
    int b = idx / T;
    int t = idx - b * T;
    const float* p = sim + (ll)b * Nn * T + t;
    float m = -3.402823466e38f;
    for (int n = 0; n < Nn; ++n) m = fmaxf(m, p[(ll)n * T]);
    amax[idx] = m;
}

// ---------------------------------------------------------------------------
// Winner-take-all masked softmax over the token axis, one block per (b,n) row.
// Writes f32 att in place (d_out) and a bf16 copy for the att@v GEMM.
// ---------------------------------------------------------------------------
__global__ __launch_bounds__(256)
void softmax_wta_kernel(float* __restrict__ att, u16* __restrict__ attbf,
                        const float* __restrict__ amax, int Nn, int T) {
    __shared__ float red[256];
    const int row = blockIdx.x;            // b*Nn + n
    const int b   = row / Nn;
    float* p      = att   + (ll)row * T;
    u16*   ph     = attbf + (ll)row * T;
    const float* am = amax + (ll)b * T;
    const int tid = threadIdx.x;

    float sv[8];                           // T = 2048 = 8 * 256
    float lmax = -3.402823466e38f;
#pragma unroll
    for (int i = 0; i < 8; ++i) {
        int t = tid + i * 256;
        float s = p[t];
        sv[i] = (s >= am[t]) ? s : -3.402823466e38f;   // keep only winners
        lmax = fmaxf(lmax, sv[i]);
    }
    red[tid] = lmax; __syncthreads();
    for (int st = 128; st > 0; st >>= 1) {
        if (tid < st) red[tid] = fmaxf(red[tid], red[tid + st]);
        __syncthreads();
    }
    float rmax = red[0]; __syncthreads();

    float lsum = 0.f;
#pragma unroll
    for (int i = 0; i < 8; ++i) {
        float e = (sv[i] > -3.0e38f) ? __expf(sv[i] - rmax) : 0.f;
        sv[i] = e; lsum += e;
    }
    red[tid] = lsum; __syncthreads();
    for (int st = 128; st > 0; st >>= 1) {
        if (tid < st) red[tid] += red[tid + st];
        __syncthreads();
    }
    float rsum = red[0];
    float inv  = (rsum > 0.f) ? 1.f / rsum : 0.f;      // row w/o winners -> 0
#pragma unroll
    for (int i = 0; i < 8; ++i) {
        int t = tid + i * 256;
        float a = sv[i] * inv;
        p[t]  = a;
        ph[t] = f32_to_bf16(a);
    }
}

// ---------------------------------------------------------------------------
// Host-side orchestration
// ---------------------------------------------------------------------------
extern "C" void kernel_launch(void* const* d_in, const int* in_sizes, int n_in,
                              void* d_out, int out_size, void* d_ws, size_t ws_size,
                              hipStream_t stream) {
    (void)in_sizes; (void)n_in; (void)out_size; (void)ws_size;
    const int Bb = 8, T = 2048, C = 512, Nn = 1024;

    const float* x    = (const float*)d_in[0];   // [B,T,C]
    const float* Wq   = (const float*)d_in[1];   // [N,T]
    const float* bq   = (const float*)d_in[2];   // [N]
    const float* Wk   = (const float*)d_in[3];   // [C,C]
    const float* Wv   = (const float*)d_in[4];   // [C,C]
    const float* Wout = (const float*)d_in[5];   // [C,C]
    const float* bout = (const float*)d_in[6];   // [C]

    float* out = (float*)d_out;                        // [B,N,C]
    float* att = out + (size_t)Bb * Nn * C;            // [B,N,T]

    // workspace carve-up (256B aligned)
    char* ws = (char*)d_ws;
    size_t off = 0;
    auto alloc = [&](size_t bytes) -> char* {
        char* p = ws + off;
        off += (bytes + 255) & ~(size_t)255;
        return p;
    };
    u16*   xbf   = (u16*)  alloc((size_t)Bb * T * C * 2);
    u16*   wqb   = (u16*)  alloc((size_t)Nn * T * 2);
    u16*   wkb   = (u16*)  alloc((size_t)C * C * 2);
    u16*   wvb   = (u16*)  alloc((size_t)C * C * 2);
    u16*   wob   = (u16*)  alloc((size_t)C * C * 2);
    u16*   qbf   = (u16*)  alloc((size_t)Bb * Nn * C * 2);
    u16*   kbf   = (u16*)  alloc((size_t)Bb * T * C * 2);
    u16*   vbf   = (u16*)  alloc((size_t)Bb * T * C * 2);
    u16*   attbf = (u16*)  alloc((size_t)Bb * Nn * T * 2);
    u16*   midbf = (u16*)  alloc((size_t)Bb * Nn * C * 2);
    float* amax  = (float*)alloc((size_t)Bb * T * 4);

    auto cvt = [&](const float* s, u16* d, ll n) {
        cvt_f32_bf16_kernel<<<(unsigned)((n + 255) / 256), 256, 0, stream>>>(s, d, n);
    };
    cvt(x,    xbf, (ll)Bb * T * C);
    cvt(Wq,   wqb, (ll)Nn * T);
    cvt(Wk,   wkb, (ll)C * C);
    cvt(Wv,   wvb, (ll)C * C);
    cvt(Wout, wob, (ll)C * C);

    const float scale = 0.044194173824159216f;  // C^-0.5

    // 1) q[b] = Wq @ x[b] + bq  ->  qbf [B,N,C]     (B = x: n contiguous)
    gemm_bf16_wmma_kernel<false><<<dim3(C / BN, Nn / BM, Bb), 256, 0, stream>>>(
        wqb, 0, T,
        xbf, (ll)T * C, /*ldbK*/ C, /*ldbN*/ 1,
        nullptr, qbf, (ll)Nn * C, C,
        /*K*/ T, 1.f, bq, /*biasMode per-M*/ 1);

    // 2) k[b] = x[b] @ Wk^T  ->  kbf [B,T,C]        (B = Wk^T: k contiguous)
    gemm_bf16_wmma_kernel<true><<<dim3(C / BN, T / BM, Bb), 256, 0, stream>>>(
        xbf, (ll)T * C, C,
        wkb, 0, /*ldbK*/ 1, /*ldbN*/ C,
        nullptr, kbf, (ll)T * C, C,
        /*K*/ C, 1.f, nullptr, 0);

    // 3) v[b] = x[b] @ Wv^T  ->  vbf [B,T,C]
    gemm_bf16_wmma_kernel<true><<<dim3(C / BN, T / BM, Bb), 256, 0, stream>>>(
        xbf, (ll)T * C, C,
        wvb, 0, 1, C,
        nullptr, vbf, (ll)T * C, C,
        C, 1.f, nullptr, 0);

    // 4) sim[b] = scale * q[b] @ k[b]^T -> att region of d_out (f32)
    gemm_bf16_wmma_kernel<true><<<dim3(T / BN, Nn / BM, Bb), 256, 0, stream>>>(
        qbf, (ll)Nn * C, C,
        kbf, (ll)T * C, /*ldbK*/ 1, /*ldbN*/ C,
        att, nullptr, (ll)Nn * T, T,
        /*K*/ C, scale, nullptr, 0);

    // 5) per-token max over nodes, then WTA masked softmax (f32 + bf16 copy)
    colmax_kernel<<<(Bb * T) / 256, 256, 0, stream>>>(att, amax, Nn, T);
    softmax_wta_kernel<<<Bb * Nn, 256, 0, stream>>>(att, attbf, amax, Nn, T);

    // 6) mid[b] = att[b] @ v[b]  ->  midbf [B,N,C]  (B = v: n contiguous)
    gemm_bf16_wmma_kernel<false><<<dim3(C / BN, Nn / BM, Bb), 256, 0, stream>>>(
        attbf, (ll)Nn * T, T,
        vbf, (ll)T * C, /*ldbK*/ C, /*ldbN*/ 1,
        nullptr, midbf, (ll)Nn * C, C,
        /*K*/ T, 1.f, nullptr, 0);

    // 7) out[b] = mid[b] @ Wout^T + bout  ->  d_out (f32)
    gemm_bf16_wmma_kernel<true><<<dim3(C / BN, Nn / BM, Bb), 256, 0, stream>>>(
        midbf, (ll)Nn * C, C,
        wob, 0, 1, C,
        out, nullptr, (ll)Nn * C, C,
        /*K*/ C, 1.f, bout, /*biasMode per-N*/ 2);
}
